// GraphConditioningModule_39024072851915
// MI455X (gfx1250) — compile-verified, matrix operand order checked by reading, not run on hardware
//
#include <hip/hip_runtime.h>

// ---------------------------------------------------------------------------
// GCN (2-layer GCNConv + mean pool) for MI455X / gfx1250, wave32 + WMMA.
//
// Pipeline:
//   deg -> dis -> norm
//   T = f16(chord_feats) @ W1            (WMMA, 1008x128x256, rows clamped)
//   agg1[c] += norm(e) * T[node_idx[row(e)]]   (f32 atomics, L2 resident)
//   x1h = f16(relu(agg1 + b1))
//   x2lin = x1h @ W2                     (WMMA, 65536x256x128)
//   agg2[c] += norm(e) * x2lin[row(e)]
//   x2 = relu(agg2 + b2)
//   out = segment_mean(x2, batch)
//
// v2: software-pipelined WMMA GEMM (double-buffered A/B fragments in
// registers) so next-kt VMEM loads overlap current-kt matrix ops instead of
// each WMMA stalling on s_wait_loadcnt 0.
// ---------------------------------------------------------------------------

typedef __attribute__((ext_vector_type(16))) _Float16 v16h;
typedef __attribute__((ext_vector_type(8)))  _Float16 v8h;
typedef __attribute__((ext_vector_type(8)))  float    v8f;

constexpr int N_NODES = 65536;
constexpr int N_EDGES = 2097152;
constexpr int VOCAB   = 1000;
constexpr int F_INC   = 128;
constexpr int HDIM    = 256;
constexpr int ODIM    = 128;
constexpr int NGRAPH  = 256;
constexpr int EP      = N_EDGES + N_NODES;   // edges + self loops
constexpr int VPAD    = 1008;                // 63 tiles of 16

// ------------------------------- utility ----------------------------------
__global__ void fill_f32(float* p, float v, int n) {
    int i = blockIdx.x * blockDim.x + threadIdx.x;
    if (i < n) p[i] = v;
}

__global__ void deg_kernel(const int* __restrict__ col,
                           const float* __restrict__ ew,
                           float* __restrict__ deg) {
    int e = blockIdx.x * blockDim.x + threadIdx.x;
    if (e < N_EDGES) atomicAdd(&deg[col[e]], ew[e]);
}

__global__ void dis_kernel(float* deg) {
    int i = blockIdx.x * blockDim.x + threadIdx.x;
    if (i < N_NODES) {
        float d = deg[i];
        deg[i] = (d > 0.0f) ? rsqrtf(d) : 0.0f;   // deg>=1 always (self loop)
    }
}

__global__ void norm_kernel(const int* __restrict__ row,
                            const int* __restrict__ col,
                            const float* __restrict__ ew,
                            const float* __restrict__ dis,
                            float* __restrict__ norm) {
    int e = blockIdx.x * blockDim.x + threadIdx.x;
    if (e >= EP) return;
    int r, c; float w;
    if (e < N_EDGES) { r = row[e]; c = col[e]; w = ew[e]; }
    else             { r = c = e - N_EDGES;     w = 1.0f; }
    norm[e] = dis[r] * w * dis[c];
}

__global__ void f32_to_f16(const float* __restrict__ src,
                           _Float16* __restrict__ dst, int n) {
    int i = blockIdx.x * blockDim.x + threadIdx.x;
    if (i < n) dst[i] = (_Float16)src[i];
}

// Pre-swizzle a row-major f32 B matrix [K x Nn] into the CDNA5 WMMA B-fragment
// layout for v_wmma_*_16x16x32: fragment (kt,nt), lane l holds column n=nt*16+
// (l&15); halves 0..7 = K (kt*32 + (l<16?0:8) + 0..7), halves 8..15 = +16.
__global__ void swizzle_b(const float* __restrict__ B,
                          _Float16* __restrict__ frag, int K, int Nn) {
    int total = K * Nn;
    int i = blockIdx.x * blockDim.x + threadIdx.x;
    if (i >= total) return;
    int h    = i & 15;
    int lane = (i >> 4) & 31;
    int t    = i >> 9;                    // fragment id = kt*ntTotal + nt
    int ntTotal = Nn >> 4;
    int kt = t / ntTotal, nt = t % ntTotal;
    int k = kt * 32 + (h & 7) + ((h >> 3) << 4) + ((lane >> 4) << 3);
    int n = nt * 16 + (lane & 15);
    frag[i] = (_Float16)B[(size_t)k * Nn + n];
}

// ------------------------------ WMMA GEMM ----------------------------------
// C[M,Nn] = A[M,K] (f16 row-major) @ Bfrag (pre-swizzled f16), f32 accumulate.
// One wave computes a 16x64 tile of C (4 accumulators). Row index clamped to
// mClamp-1 so M can be padded to a multiple of 16 (VOCAB=1000 -> 1008).
// Software-pipelined: fragments for kt+1 are loaded before the kt WMMAs are
// issued, so VMEM latency overlaps matrix ops.

__device__ __forceinline__ v16h load_a_frag(const _Float16* __restrict__ arow,
                                            int kt) {
    v8h lo = *reinterpret_cast<const v8h*>(arow + kt * 32);       // K+0..7
    v8h hi = *reinterpret_cast<const v8h*>(arow + kt * 32 + 16);  // K+16..23
    v16h a;
#pragma unroll
    for (int i = 0; i < 8; ++i) { a[i] = lo[i]; a[i + 8] = hi[i]; }
    return a;
}

__global__ void wmma_gemm(const _Float16* __restrict__ A,
                          const _Float16* __restrict__ Bfrag,
                          float* __restrict__ C,
                          int Mtiles, int K, int Nn, int mClamp) {
    int ntTotal = Nn >> 4;
    int nquads  = Nn >> 6;                  // 64 columns per wave
    int wave = (blockIdx.x * blockDim.x + threadIdx.x) >> 5;
    int lane = threadIdx.x & 31;
    int mtile = wave / nquads;
    int nquad = wave % nquads;
    if (mtile >= Mtiles) return;            // whole-wave exit: EXEC stays full

    int m = mtile * 16 + (lane & 15);
    if (m >= mClamp) m = mClamp - 1;
    // A-fragment lane layout: lanes 0-15 start at K=0, lanes 16-31 at K=8.
    const _Float16* arow = A + (size_t)m * K + ((lane >> 4) << 3);
    const v16h* bbase = reinterpret_cast<const v16h*>(Bfrag)
                      + (size_t)(nquad * 4) * 32 + lane;

    v8f acc0 = {}, acc1 = {}, acc2 = {}, acc3 = {};
    int ktiles = K >> 5;

    // Prologue: fragments for kt = 0.
    v16h a  = load_a_frag(arow, 0);
    v16h b0 = bbase[0], b1 = bbase[32], b2 = bbase[64], b3 = bbase[96];

    for (int kt = 0; kt + 1 < ktiles; ++kt) {
        // Prefetch next iteration's fragments into fresh registers.
        v16h an = load_a_frag(arow, kt + 1);
        const v16h* bn = bbase + (size_t)(kt + 1) * ntTotal * 32;
        v16h bn0 = bn[0], bn1 = bn[32], bn2 = bn[64], bn3 = bn[96];

        acc0 = __builtin_amdgcn_wmma_f32_16x16x32_f16(false, a, false, b0,
                                                      (short)0, acc0, false, false);
        acc1 = __builtin_amdgcn_wmma_f32_16x16x32_f16(false, a, false, b1,
                                                      (short)0, acc1, false, false);
        acc2 = __builtin_amdgcn_wmma_f32_16x16x32_f16(false, a, false, b2,
                                                      (short)0, acc2, false, false);
        acc3 = __builtin_amdgcn_wmma_f32_16x16x32_f16(false, a, false, b3,
                                                      (short)0, acc3, false, false);

        a = an; b0 = bn0; b1 = bn1; b2 = bn2; b3 = bn3;
    }

    // Epilogue: last kt.
    acc0 = __builtin_amdgcn_wmma_f32_16x16x32_f16(false, a, false, b0,
                                                  (short)0, acc0, false, false);
    acc1 = __builtin_amdgcn_wmma_f32_16x16x32_f16(false, a, false, b1,
                                                  (short)0, acc1, false, false);
    acc2 = __builtin_amdgcn_wmma_f32_16x16x32_f16(false, a, false, b2,
                                                  (short)0, acc2, false, false);
    acc3 = __builtin_amdgcn_wmma_f32_16x16x32_f16(false, a, false, b3,
                                                  (short)0, acc3, false, false);

    // C 16x16 f32 layout: VGPR r -> row r (lanes 0-15) / row 8+r (lanes 16-31)
    int rowBase = mtile * 16 + ((lane >> 4) << 3);
    int colBase = nquad * 64 + (lane & 15);
#pragma unroll
    for (int r = 0; r < 8; ++r) {
        float* crow = C + (size_t)(rowBase + r) * Nn + colBase;
        crow[0]  = acc0[r];
        crow[16] = acc1[r];
        crow[32] = acc2[r];
        crow[48] = acc3[r];
    }
}

// --------------------------- edge aggregation ------------------------------
// dst[col(e)] += norm(e) * src[srcRowOf ? srcRowOf[row(e)] : row(e)]
// Thread = (edge, 4-feature chunk). All operands L2-resident.
__global__ void aggregate_kernel(const float* __restrict__ src,
                                 const int* __restrict__ srcRowOf,
                                 const int* __restrict__ row,
                                 const int* __restrict__ col,
                                 const float* __restrict__ norm,
                                 float* __restrict__ dst,
                                 int F, int fqShift) {
    long long t = (long long)blockIdx.x * blockDim.x + threadIdx.x;
    long long total = (long long)EP << fqShift;     // EP * (F/4)
    if (t >= total) return;
    int e = (int)(t >> fqShift);
    int f = (int)((t & ((1 << fqShift) - 1)) << 2);
    int r, c;
    if (e < N_EDGES) { r = row[e]; c = col[e]; }
    else             { r = c = e - N_EDGES; }
    float w = norm[e];
    int sr = srcRowOf ? srcRowOf[r] : r;
    const float4 v = *reinterpret_cast<const float4*>(src + (size_t)sr * F + f);
    float* d = dst + (size_t)c * F + f;
    atomicAdd(d + 0, w * v.x);
    atomicAdd(d + 1, w * v.y);
    atomicAdd(d + 2, w * v.z);
    atomicAdd(d + 3, w * v.w);
}

__global__ void bias_relu_f16(const float* __restrict__ agg,
                              const float* __restrict__ bias,
                              _Float16* __restrict__ out,
                              int fMask, long long total) {
    long long i = (long long)blockIdx.x * blockDim.x + threadIdx.x;
    if (i >= total) return;
    float v = agg[i] + bias[(int)(i & fMask)];
    out[i] = (_Float16)fmaxf(v, 0.0f);
}

__global__ void bias_relu_f32(float* __restrict__ agg,
                              const float* __restrict__ bias,
                              int fMask, long long total) {
    long long i = (long long)blockIdx.x * blockDim.x + threadIdx.x;
    if (i >= total) return;
    agg[i] = fmaxf(agg[i] + bias[(int)(i & fMask)], 0.0f);
}

// ------------------------------- pooling -----------------------------------
__global__ void pool_kernel(const float* __restrict__ x2,
                            const int* __restrict__ batch,
                            float* __restrict__ pool,
                            float* __restrict__ cnt) {
    long long i = (long long)blockIdx.x * blockDim.x + threadIdx.x;
    long long total = (long long)N_NODES * ODIM;
    if (i >= total) return;
    int node = (int)(i >> 7);               // ODIM = 128
    int f    = (int)(i & 127);
    int b = batch[node];
    atomicAdd(&pool[(size_t)b * ODIM + f], x2[i]);
    if (f == 0) atomicAdd(&cnt[b], 1.0f);
}

__global__ void finalize_kernel(const float* __restrict__ pool,
                                const float* __restrict__ cnt,
                                float* __restrict__ out) {
    int i = blockIdx.x * blockDim.x + threadIdx.x;
    if (i >= NGRAPH * ODIM) return;
    out[i] = pool[i] / fmaxf(cnt[i >> 7], 1.0f);
}

// ------------------------------- launcher ----------------------------------
static inline unsigned blocks_for(long long total, int bs) {
    return (unsigned)((total + bs - 1) / bs);
}

extern "C" void kernel_launch(void* const* d_in, const int* in_sizes, int n_in,
                              void* d_out, int out_size, void* d_ws, size_t ws_size,
                              hipStream_t stream) {
    const float* chord = (const float*)d_in[0];
    const float* W1    = (const float*)d_in[1];
    const float* b1    = (const float*)d_in[2];
    const float* W2    = (const float*)d_in[3];
    const float* b2    = (const float*)d_in[4];
    const float* ea    = (const float*)d_in[5];
    const int*   nidx  = (const int*)  d_in[6];
    const int*   row   = (const int*)  d_in[7];            // edge_index[0]
    const int*   col   = row + N_EDGES;                    // edge_index[1]
    const int*   batch = (const int*)  d_in[8];
    float* out = (float*)d_out;
    (void)in_sizes; (void)n_in; (void)out_size; (void)ws_size;

    // --- workspace carve-up (256B aligned bumps) ---
    char* w = (char*)d_ws;
    size_t off = 0;
    auto carve = [&](size_t bytes) -> void* {
        void* p = w + off;
        off = (off + bytes + 255) & ~(size_t)255;
        return p;
    };
    float*    deg    = (float*)   carve((size_t)N_NODES * 4);
    float*    norm   = (float*)   carve((size_t)EP * 4);
    _Float16* chf16  = (_Float16*)carve((size_t)VOCAB * F_INC * 2);
    _Float16* W1frag = (_Float16*)carve((size_t)F_INC * HDIM * 2);
    _Float16* W2frag = (_Float16*)carve((size_t)HDIM * ODIM * 2);
    float*    T      = (float*)   carve((size_t)VPAD * HDIM * 4);
    float*    agg1   = (float*)   carve((size_t)N_NODES * HDIM * 4);
    _Float16* x1h    = (_Float16*)carve((size_t)N_NODES * HDIM * 2);
    float*    x2lin  = (float*)   carve((size_t)N_NODES * ODIM * 4);
    float*    agg2   = (float*)   carve((size_t)N_NODES * ODIM * 4);
    float*    pool   = (float*)   carve((size_t)NGRAPH * ODIM * 4);
    float*    cnt    = (float*)   carve((size_t)NGRAPH * 4);

    const int BS = 256;

    // 1) degrees (self-loop weight 1 baked into the init)
    fill_f32<<<blocks_for(N_NODES, BS), BS, 0, stream>>>(deg, 1.0f, N_NODES);
    deg_kernel<<<blocks_for(N_EDGES, BS), BS, 0, stream>>>(col, ea, deg);
    dis_kernel<<<blocks_for(N_NODES, BS), BS, 0, stream>>>(deg);
    norm_kernel<<<blocks_for(EP, BS), BS, 0, stream>>>(row, col, ea, deg, norm);

    // 2) f16 conversions + B-fragment swizzles (one-time, tiny)
    f32_to_f16<<<blocks_for((long long)VOCAB * F_INC, BS), BS, 0, stream>>>(
        chord, chf16, VOCAB * F_INC);
    swizzle_b<<<blocks_for((long long)F_INC * HDIM, BS), BS, 0, stream>>>(
        W1, W1frag, F_INC, HDIM);
    swizzle_b<<<blocks_for((long long)HDIM * ODIM, BS), BS, 0, stream>>>(
        W2, W2frag, HDIM, ODIM);

    // 3) T = chord_f16 @ W1  (vocab-level GEMM: 1008x128x256, WMMA)
    {
        int Mtiles = VPAD / 16, nquads = HDIM / 64;
        long long threads = (long long)Mtiles * nquads * 32;
        wmma_gemm<<<blocks_for(threads, BS), BS, 0, stream>>>(
            chf16, W1frag, T, Mtiles, F_INC, HDIM, VOCAB);
    }

    // 4) layer-1 aggregation: agg1[c] += norm * T[node_idx[r]]
    hipMemsetAsync(agg1, 0, (size_t)N_NODES * HDIM * 4, stream);
    aggregate_kernel<<<blocks_for((long long)EP * (HDIM / 4), BS), BS, 0, stream>>>(
        T, nidx, row, col, norm, agg1, HDIM, 6);
    bias_relu_f16<<<blocks_for((long long)N_NODES * HDIM, BS), BS, 0, stream>>>(
        agg1, b1, x1h, HDIM - 1, (long long)N_NODES * HDIM);

    // 5) x2lin = x1h @ W2  (65536x256x128, WMMA)
    {
        int Mtiles = N_NODES / 16, nquads = ODIM / 64;
        long long threads = (long long)Mtiles * nquads * 32;
        wmma_gemm<<<blocks_for(threads, BS), BS, 0, stream>>>(
            x1h, W2frag, x2lin, Mtiles, HDIM, ODIM, N_NODES);
    }

    // 6) layer-2 aggregation + bias/relu (in place)
    hipMemsetAsync(agg2, 0, (size_t)N_NODES * ODIM * 4, stream);
    aggregate_kernel<<<blocks_for((long long)EP * (ODIM / 4), BS), BS, 0, stream>>>(
        x2lin, nullptr, row, col, norm, agg2, ODIM, 5);
    bias_relu_f32<<<blocks_for((long long)N_NODES * ODIM, BS), BS, 0, stream>>>(
        agg2, b2, ODIM - 1, (long long)N_NODES * ODIM);

    // 7) global mean pool
    hipMemsetAsync(pool, 0, (size_t)NGRAPH * ODIM * 4, stream);
    hipMemsetAsync(cnt, 0, (size_t)NGRAPH * 4, stream);
    pool_kernel<<<blocks_for((long long)N_NODES * ODIM, BS), BS, 0, stream>>>(
        agg2, batch, pool, cnt);
    finalize_kernel<<<blocks_for(NGRAPH * ODIM, BS), BS, 0, stream>>>(pool, cnt, out);
}
